// Decoder_10359461118619
// MI455X (gfx1250) — matrix operand order repaired and checked by reading, not compiled, not run on hardware
//
#include <hip/hip_runtime.h>
#include <hip/hip_bf16.h>

// ---------------------------------------------------------------------------
// Pointer-generator LSTM decoder for MI455X (gfx1250, wave32, WMMA, TDM).
//  * all GEMMs: v_wmma_f32_16x16x32_bf16, K16-blocked B layout -> 32B B-frag loads
//  * GEMM A tiles (bf16, contiguous 64KB) moved into LDS by the Tensor Data
//    Mover (tensor_load_to_lds + s_wait_tensorcnt), B stream prefetched
//  * scan: 16 WGs (one WGP each), 3 global syncs/step; s_cluster_barrier fast
//    path + monotonic atomic barrier fallback; per-batch attention local to WG
//  * big FC (55.7 GFLOP): 64x256 blocking, W_fc bf16 resident in 192MB L2
// ---------------------------------------------------------------------------

typedef __attribute__((ext_vector_type(16))) __bf16 v16bf;
typedef __attribute__((ext_vector_type(8)))  __bf16 v8bf;
typedef __attribute__((ext_vector_type(8)))  float  v8f;
typedef __attribute__((ext_vector_type(4)))  unsigned int v4u;
typedef __attribute__((ext_vector_type(8)))  int v8i;
typedef __attribute__((ext_vector_type(4)))  int v4i;

#define B_    16
#define T_    40
#define S_    120
#define U_    512
#define E_    256
#define V_    85000
#define OOV_  50
#define VEXT_ 85050
#define KZ_   1280     // x(256) + attn(512) + h(512)
#define NZ_   2048     // 4 gates * U
#define NFCP_ 85008    // V padded to 16

__device__ __forceinline__ float sigf(float x) { return 1.f / (1.f + __expf(-x)); }

__device__ __forceinline__ v8f wmma_bf16(v16bf a, v16bf b, v8f c) {
  return __builtin_amdgcn_wmma_f32_16x16x32_bf16(false, a, false, b, (short)0, c, false, false);
}

// A-fragment (16-bit A 16x32, contiguous k = 16*half + s): lane holds two 8-elem contiguous
// chunks at k0+16*half and k0+16*half+8 of its M-row.
__device__ __forceinline__ v16bf make_frag(const __bf16* p0, const __bf16* p1) {
  union { v16bf v; v8bf h[2]; } u;
  u.h[0] = *(const v8bf*)p0;
  u.h[1] = *(const v8bf*)p1;
  return u.v;
}

// Row pointer into the virtual z-GEMM A matrix [x_t | attn | h] (all bf16).
__device__ __forceinline__ const __bf16* zrow_ptr(const __bf16* xb, const __bf16* ab,
                                                  const __bf16* hb, int m, int t, int k) {
  if (k < E_)      return xb + ((size_t)(m * T_ + t)) * E_ + k;
  if (k < E_ + U_) return ab + m * U_ + (k - E_);
  return hb + m * U_ + (k - E_ - U_);
}

// TDM: DMA `elems` bf16 elements (contiguous) from global -> LDS.
// D# built per CDNA5 ISA ch.8: group0 = {count=1 | lds_addr | global_addr | type=2},
// group1 = {data_size=2B, tensor_dim0=elems, tensor_dim1=1, tile_dim0=elems}.
// Groups 2/3 and the aux group are zero (<=2D tensor). TENSORcnt tracks completion.
__device__ __forceinline__ void tdm_load_to_lds(const void* gsrc, void* lds, unsigned elems) {
  unsigned long long ga = (unsigned long long)(uintptr_t)gsrc;
  unsigned la = (unsigned)(uintptr_t)lds;   // LDS aperture: addr[31:0] = LDS byte offset
  v4u g0 = { 1u,                                              // count=1 (valid user D#)
             la,                                              // lds_addr  [63:32]
             (unsigned)ga,                                    // global_addr[95:64]
             (unsigned)((ga >> 32) & 0x01FFFFFFu) | 0x80000000u }; // gaddr[120:96] | type=2
  v8i g1 = { (int)0x00010000,                                 // data_size=1 (2 bytes)
             (int)((elems & 0xFFFFu) << 16),                  // tensor_dim0[15:0] @ bits 63:48
             (int)(((elems >> 16) & 0xFFFFu) | (1u << 16)),   // tensor_dim0[31:16] | tensor_dim1=1
             (int)((elems & 0xFFFFu) << 16),                  // tile_dim0 @ bits 127:112
             0,                                               // tile_dim1=0(unused), tile_dim2=0
             (int)elems,                                      // tensor_dim0_stride low32
             0, 0 };
  v4i z4 = {};
  v8i z8 = {};
  __builtin_amdgcn_tensor_load_to_lds(g0, g1, z4, z4, z8, 0);
  __builtin_amdgcn_s_wait_tensorcnt(0);
}

// ---------------------------------------------------------------------------
// Prep: f32 -> bf16, K16-blocked weight layout Wt[(k>>4)][n][k&15]
// ---------------------------------------------------------------------------
__global__ void convert_blockK(const float* __restrict__ W, __bf16* __restrict__ Wt,
                               int K, int N, int Npad) {
  size_t idx = (size_t)blockIdx.x * 256 + threadIdx.x;
  size_t total = (size_t)K * Npad;
  if (idx >= total) return;
  int k = (int)(idx / Npad);
  int n = (int)(idx % Npad);
  float v = (n < N) ? W[(size_t)k * N + n] : 0.f;
  Wt[((size_t)(k >> 4) * Npad + n) * 16 + (k & 15)] = (__bf16)v;
}

// Wz = [Wi ; Wh] rows (1280 x 2048), K16-blocked bf16
__global__ void build_wz(const float* __restrict__ Wi, const float* __restrict__ Wh,
                         __bf16* __restrict__ Wt) {
  size_t idx = (size_t)blockIdx.x * 256 + threadIdx.x;
  if (idx >= (size_t)KZ_ * NZ_) return;
  int k = (int)(idx >> 11), n = (int)(idx & 2047);
  float v = (k < E_ + U_) ? Wi[(size_t)k * NZ_ + n] : Wh[(size_t)(k - E_ - U_) * NZ_ + n];
  Wt[((size_t)(k >> 4) * NZ_ + n) * 16 + (k & 15)] = (__bf16)v;
}

// plain f32 -> bf16 copy (row-major)
__global__ void convert_bf16(const float* __restrict__ src, __bf16* __restrict__ dst, int n) {
  int idx = blockIdx.x * 256 + threadIdx.x;
  if (idx < n) dst[idx] = (__bf16)src[idx];
}

// embedding gather -> bf16 x[B,T,E]
__global__ void gather_x(const int* __restrict__ dec, const float* __restrict__ emb,
                         __bf16* __restrict__ xbf) {
  int idx = blockIdx.x * 256 + threadIdx.x;
  if (idx >= B_ * T_ * E_) return;
  int e = idx & (E_ - 1), bt = idx >> 8;
  xbf[idx] = (__bf16)emb[(size_t)dec[bt] * E_ + e];
}

// ---------------------------------------------------------------------------
// Tiled WMMA GEMM: out[M,out_ld] = rowscale * (A_bf16[M,K] @ B + bias)
// B is K16-blocked bf16 [K/16][Npad][16]. Block = 64 rows x 256 cols,
// 256 threads (8 waves), each wave 2 n-tiles x 4 m-tiles. M % 64 == 0.
// A tile (contiguous 64*K bf16) is DMA'd into LDS by the TDM.
// ---------------------------------------------------------------------------
__global__ void __launch_bounds__(256) gemm_tiled(
    const __bf16* __restrict__ Abf, const __bf16* __restrict__ Bt,
    float* __restrict__ out, int M, int K, int N, int Npad, int out_ld,
    const float* __restrict__ biasv, const float* __restrict__ rowscale) {
  extern __shared__ __bf16 As[];  // [64][K] bf16
  const int tid = threadIdx.x, lane = tid & 31, wave = tid >> 5;
  const int half = lane >> 4, ln15 = lane & 15;
  const int m0 = blockIdx.y * 64;
  const int nt0 = blockIdx.x * 16;

  if (wave == 0)  // one DMA per block; other waves held at the barrier
    tdm_load_to_lds(Abf + (size_t)m0 * K, As, (unsigned)(64 * K));
  __syncthreads();

  const bool valid0 = (nt0 + wave * 2 + 0) * 16 < Npad;
  const bool valid1 = (nt0 + wave * 2 + 1) * 16 < Npad;
  v8f zero = {};
  v8f acc[4][2];
#pragma unroll
  for (int mt = 0; mt < 4; ++mt) { acc[mt][0] = zero; acc[mt][1] = zero; }

  const int ksteps = K >> 5;
  for (int ks = 0; ks < ksteps; ++ks) {
    int k0 = ks * 32;
    v16bf a[4];
#pragma unroll
    for (int mt = 0; mt < 4; ++mt) {
      const __bf16* p = As + (size_t)(mt * 16 + ln15) * K + k0 + 16 * half;
      a[mt] = make_frag(p, p + 8);
    }
#pragma unroll
    for (int j = 0; j < 2; ++j) {
      if (j == 0 ? !valid0 : !valid1) continue;
      int ncol = (nt0 + wave * 2 + j) * 16 + ln15;
      v16bf bb = *(const v16bf*)(Bt + ((size_t)(2 * ks + half) * Npad + ncol) * 16);
      if (ks + 1 < ksteps)  // stream W one k-step ahead (global_prefetch_b8)
        __builtin_prefetch(Bt + ((size_t)(2 * ks + 2 + half) * Npad + ncol) * 16, 0, 1);
#pragma unroll
      for (int mt = 0; mt < 4; ++mt) acc[mt][j] = wmma_bf16(a[mt], bb, acc[mt][j]);
    }
  }

#pragma unroll
  for (int j = 0; j < 2; ++j) {
    if (j == 0 ? !valid0 : !valid1) continue;
    int ncol = (nt0 + wave * 2 + j) * 16 + ln15;
    if (ncol >= N) continue;
    float bv = biasv ? biasv[ncol] : 0.f;
#pragma unroll
    for (int mt = 0; mt < 4; ++mt)
#pragma unroll
      for (int v = 0; v < 8; ++v) {
        int m = m0 + mt * 16 + v + 8 * half;
        float val = acc[mt][j][v] + bv;
        if (rowscale) val *= rowscale[m];
        out[(size_t)m * out_ld + ncol] = val;
      }
  }
}

// ---------------------------------------------------------------------------
// Recurrent scan: 16 workgroups x 1024 threads. WG w owns u-columns
// [w*32, w*32+32) of z/gates/c, and is the sole owner of batch row w for
// score/softmax/context. 3 device-wide syncs per step.
// ---------------------------------------------------------------------------
__global__ void __launch_bounds__(1024) scan_kernel(
    const __bf16* __restrict__ xbf, const __bf16* __restrict__ Wz_t,
    const __bf16* __restrict__ Wa_t, const float* __restrict__ bias,
    const float* __restrict__ keys, const float* __restrict__ memory,
    const float* __restrict__ init_h, const float* __restrict__ init_c,
    __bf16* __restrict__ h_bf /*[2][B][U] double-buffered*/,
    __bf16* __restrict__ ctx_bf, __bf16* __restrict__ attn_bf,
    float* __restrict__ hs, float* __restrict__ cs,
    float* __restrict__ attns, __bf16* __restrict__ attns_bf,
    float* __restrict__ aligns, unsigned* __restrict__ bar) {
  __shared__ float zt[4 * 512];   // [gate][b*32+ui]
  __shared__ float cst[512];      // persistent c for this WG's 32 columns
  __shared__ float red8[1024];
  __shared__ float scoreL[128];
  __shared__ float alignL[128];
  __shared__ float redC[1024];
  __shared__ float atile[512];

  const int wg = blockIdx.x;          // also the batch row this WG owns
  const int tid = threadIdx.x;
  const int lane = tid & 31, wave = tid >> 5;
  const int half = lane >> 4, ln15 = lane & 15;

  unsigned bar_epoch = 0;
  auto gbar = [&]() {
    __syncthreads();
    if (wave == 0) __builtin_amdgcn_s_cluster_barrier();  // HW fast path if clustered; NOP otherwise
    ++bar_epoch;
    if (tid == 0) {
      __threadfence();
      atomicAdd(bar, 1u);
      unsigned target = 16u * bar_epoch;
      while (atomicAdd(bar, 0u) < target) __builtin_amdgcn_s_sleep(8);
    }
    __syncthreads();
    __threadfence();
  };

  // init: c slice in LDS, h0/attn0 slices in global
  for (int i = tid; i < 512; i += 1024) {
    int b = i >> 5, ui = i & 31, u = wg * 32 + ui;
    cst[i] = init_c[b * U_ + u];
    h_bf[b * U_ + u] = (__bf16)init_h[b * U_ + u];   // buffer 0 = h at t=-1
    attn_bf[b * U_ + u] = (__bf16)0.f;
  }
  gbar();

  for (int t = 0; t < T_; ++t) {
    const __bf16* h_prev = h_bf + (size_t)(t & 1) * B_ * U_;
    __bf16* h_cur = h_bf + (size_t)((t + 1) & 1) * B_ * U_;

    // ---- P1: z = [x_t|attn|h] @ Wz, gates, h/c update -------------------
    for (int i = tid; i < 4 * 512; i += 1024) zt[i] = 0.f;
    __syncthreads();
    {
      const int combo = wave >> 2;       // (tile, gate)
      const int tile = combo >> 2;       // 0..1
      const int gate = combo & 3;        // 0..3
      const int kq = wave & 3;           // split-K quarter (320 each)
      const int ncol = gate * U_ + wg * 32 + tile * 16 + ln15;
      v8f acc = {};
      int k0 = kq * 320;
#pragma unroll
      for (int ks = 0; ks < 10; ++ks, k0 += 32) {
        const __bf16* p0 = zrow_ptr(xbf, attn_bf, h_prev, ln15, t, k0 + 16 * half);
        const __bf16* p1 = zrow_ptr(xbf, attn_bf, h_prev, ln15, t, k0 + 16 * half + 8);
        v16bf a = make_frag(p0, p1);
        v16bf bb = *(const v16bf*)(Wz_t + ((size_t)((k0 >> 4) + half) * NZ_ + ncol) * 16);
        acc = wmma_bf16(a, bb, acc);
      }
#pragma unroll
      for (int v = 0; v < 8; ++v)
        atomicAdd(&zt[gate * 512 + (v + 8 * half) * 32 + tile * 16 + ln15], acc[v]);
    }
    __syncthreads();
    if (tid < 512) {
      int b = tid >> 5, ui = tid & 31, u = wg * 32 + ui;
      float zi = zt[tid] + bias[u];
      float zf = zt[512 + tid] + bias[U_ + u];
      float zg = zt[1024 + tid] + bias[2 * U_ + u];
      float zo = zt[1536 + tid] + bias[3 * U_ + u];
      float c = sigf(zf) * cst[tid] + sigf(zi) * tanhf(zg);
      float h = sigf(zo) * tanhf(c);
      cst[tid] = c;
      h_cur[b * U_ + u] = (__bf16)h;
      hs[((size_t)(b * T_ + t)) * U_ + u] = h;
      cs[((size_t)(b * T_ + t)) * U_ + u] = c;
    }
    gbar();

    // ---- P2 (local to WG wg == batch row): score, softmax, context ------
    {
      int pr = tid >> 3, q = tid & 7;
      float part = 0.f;
      if (pr < S_) {
        const float* kp = keys + ((size_t)(wg * S_ + pr)) * U_ + q * 64;
        const __bf16* hp = h_cur + wg * U_ + q * 64;
#pragma unroll 4
        for (int u = 0; u < 64; ++u) part += (float)hp[u] * kp[u];
      }
      red8[tid] = part;
    }
    __syncthreads();
    if (tid < S_) {
      float s = 0.f;
#pragma unroll
      for (int j = 0; j < 8; ++j) s += red8[tid * 8 + j];
      scoreL[tid] = s;
    }
    __syncthreads();
    if (tid < 32) {
      float mx = -3.4e38f;
      for (int s = tid; s < S_; s += 32) mx = fmaxf(mx, scoreL[s]);
      for (int off = 16; off; off >>= 1) mx = fmaxf(mx, __shfl_xor(mx, off, 32));
      float sum = 0.f;
      for (int s = tid; s < S_; s += 32) { float e = __expf(scoreL[s] - mx); scoreL[s] = e; sum += e; }
      for (int off = 16; off; off >>= 1) sum += __shfl_xor(sum, off, 32);
      float inv = 1.f / sum;
      for (int s = tid; s < S_; s += 32) {
        float a = scoreL[s] * inv;
        alignL[s] = a;
        aligns[((size_t)(wg * T_ + t)) * S_ + s] = a;
      }
    }
    __syncthreads();
    {
      int u = tid >> 1, hh = tid & 1;
      const float* mp = memory + (size_t)wg * S_ * U_ + u;
      float part = 0.f;
      for (int s = hh * 60; s < hh * 60 + 60; ++s) part += alignL[s] * mp[(size_t)s * U_];
      redC[tid] = part;
    }
    __syncthreads();
    if (tid < U_) ctx_bf[wg * U_ + tid] = (__bf16)(redC[tid * 2] + redC[tid * 2 + 1]);
    gbar();

    // ---- P3: attn = [h|ctx] @ W_attn (this WG's 32 columns, all rows) ---
    for (int i = tid; i < 512; i += 1024) atile[i] = 0.f;
    __syncthreads();
    {
      const int nt = wave & 1;
      const int kq = wave >> 1;  // 16 split-K chunks of 64
      const int ncol = wg * 32 + nt * 16 + ln15;
      v8f acc = {};
      int k0 = kq * 64;
#pragma unroll
      for (int ks = 0; ks < 2; ++ks, k0 += 32) {
        int ka = k0 + 16 * half, kb = k0 + 16 * half + 8;
        const __bf16* p0 = (ka < U_) ? (h_cur + ln15 * U_ + ka) : (ctx_bf + ln15 * U_ + ka - U_);
        const __bf16* p1 = (kb < U_) ? (h_cur + ln15 * U_ + kb) : (ctx_bf + ln15 * U_ + kb - U_);
        v16bf a = make_frag(p0, p1);
        v16bf bb = *(const v16bf*)(Wa_t + ((size_t)((k0 >> 4) + half) * U_ + ncol) * 16);
        acc = wmma_bf16(a, bb, acc);
      }
#pragma unroll
      for (int v = 0; v < 8; ++v)
        atomicAdd(&atile[(v + 8 * half) * 32 + nt * 16 + ln15], acc[v]);
    }
    __syncthreads();
    if (tid < 512) {
      int b = tid >> 5, ui = tid & 31, u = wg * 32 + ui;
      float a = atile[tid];
      attns[((size_t)(b * T_ + t)) * U_ + u] = a;
      attns_bf[((size_t)(b * T_ + t)) * U_ + u] = (__bf16)a;  // FC A-operand
      attn_bf[b * U_ + u] = (__bf16)a;
    }
    gbar();
  }
}

// ---------------------------------------------------------------------------
// p_gen[r] = sigmoid( [h|c|x|attn] . W_g + b_g ), one wave per row
// ---------------------------------------------------------------------------
__global__ void __launch_bounds__(256) pgen_kernel(
    const float* __restrict__ hs, const float* __restrict__ cs,
    const float* __restrict__ attns, const int* __restrict__ dec,
    const float* __restrict__ emb, const float* __restrict__ Wg,
    const float* __restrict__ bg, float* __restrict__ pgen) {
  int row = blockIdx.x * 8 + (threadIdx.x >> 5);
  int lane = threadIdx.x & 31;
  if (row >= B_ * T_) return;
  const float* h = hs + (size_t)row * U_;
  const float* c = cs + (size_t)row * U_;
  const float* a = attns + (size_t)row * U_;
  float acc = 0.f;
  for (int u = lane; u < U_; u += 32)
    acc += h[u] * Wg[u] + c[u] * Wg[U_ + u] + a[u] * Wg[2 * U_ + E_ + u];
  const float* x = emb + (size_t)dec[row] * E_;
  for (int e = lane; e < E_; e += 32) acc += x[e] * Wg[2 * U_ + e];
  for (int off = 16; off; off >>= 1) acc += __shfl_xor(acc, off, 32);
  if (lane == 0) pgen[row] = sigf(acc + bg[0]);
}

__global__ void zero_oov(float* __restrict__ out) {
  int idx = blockIdx.x * 256 + threadIdx.x;
  if (idx >= B_ * T_ * OOV_) return;
  int r = idx / OOV_, j = idx - r * OOV_;
  out[(size_t)r * VEXT_ + V_ + j] = 0.f;
}

__global__ void scatter_kernel(const int* __restrict__ eids, const float* __restrict__ aligns,
                               const float* __restrict__ pgen, float* __restrict__ out) {
  int idx = blockIdx.x * 256 + threadIdx.x;
  if (idx >= B_ * T_ * S_) return;
  int r = idx / S_, s = idx - r * S_;
  int id = eids[(r / T_) * S_ + s];
  float val = (1.f - pgen[r]) * aligns[(size_t)r * S_ + s];
  atomicAdd(&out[(size_t)r * VEXT_ + id], val);
}

// ---------------------------------------------------------------------------
extern "C" void kernel_launch(void* const* d_in, const int* in_sizes, int n_in,
                              void* d_out, int out_size, void* d_ws, size_t ws_size,
                              hipStream_t stream) {
  const int*   dec   = (const int*)d_in[0];
  const float* mem   = (const float*)d_in[1];
  const float* ih    = (const float*)d_in[2];
  const float* ic    = (const float*)d_in[3];
  const int*   eids  = (const int*)d_in[4];
  const float* emb   = (const float*)d_in[5];
  const float* Wk    = (const float*)d_in[6];
  const float* Wi    = (const float*)d_in[7];
  const float* Wh    = (const float*)d_in[8];
  const float* bz    = (const float*)d_in[9];
  const float* Wattn = (const float*)d_in[10];
  const float* Wfc   = (const float*)d_in[11];
  const float* bfc   = (const float*)d_in[12];
  const float* Wg    = (const float*)d_in[13];
  const float* bg    = (const float*)d_in[14];
  float* out = (float*)d_out;

  char* p = (char*)d_ws;
  auto take = [&](size_t bytes) { char* r = p; p += (bytes + 255) & ~(size_t)255; return r; };
  __bf16* Wfc_t    = (__bf16*)take((size_t)(U_ / 16) * NFCP_ * 16 * 2);   // 87 MB (L2-resident)
  __bf16* Wz_t     = (__bf16*)take((size_t)(KZ_ / 16) * NZ_ * 16 * 2);
  __bf16* Wa_t     = (__bf16*)take((size_t)(2 * U_ / 16) * U_ * 16 * 2);
  __bf16* Wk_t     = (__bf16*)take((size_t)(U_ / 16) * U_ * 16 * 2);
  __bf16* xbf      = (__bf16*)take((size_t)B_ * T_ * E_ * 2);
  __bf16* mem_bf   = (__bf16*)take((size_t)B_ * S_ * U_ * 2);
  float*  keys     = (float*)take((size_t)B_ * S_ * U_ * 4);
  float*  hs       = (float*)take((size_t)B_ * T_ * U_ * 4);
  float*  cs       = (float*)take((size_t)B_ * T_ * U_ * 4);
  float*  attns    = (float*)take((size_t)B_ * T_ * U_ * 4);
  __bf16* attns_bf = (__bf16*)take((size_t)B_ * T_ * U_ * 2);
  float*  aligns   = (float*)take((size_t)B_ * T_ * S_ * 4);
  float*  pgen     = (float*)take((size_t)B_ * T_ * 4);
  __bf16* h_bf     = (__bf16*)take((size_t)2 * B_ * U_ * 2);
  __bf16* ctx_bf   = (__bf16*)take((size_t)B_ * U_ * 2);
  __bf16* attn_bf  = (__bf16*)take((size_t)B_ * U_ * 2);
  unsigned* bar    = (unsigned*)take(16);

  // --- prep: conversions / packing ---
  gather_x<<<(B_ * T_ * E_ + 255) / 256, 256, 0, stream>>>(dec, emb, xbf);
  build_wz<<<(KZ_ * NZ_ + 255) / 256, 256, 0, stream>>>(Wi, Wh, Wz_t);
  convert_blockK<<<(2 * U_ * U_ + 255) / 256, 256, 0, stream>>>(Wattn, Wa_t, 2 * U_, U_, U_);
  convert_blockK<<<(U_ * U_ + 255) / 256, 256, 0, stream>>>(Wk, Wk_t, U_, U_, U_);
  {
    size_t total = (size_t)U_ * NFCP_;
    convert_blockK<<<(unsigned)((total + 255) / 256), 256, 0, stream>>>(Wfc, Wfc_t, U_, V_, NFCP_);
  }
  convert_bf16<<<(B_ * S_ * U_ + 255) / 256, 256, 0, stream>>>(mem, mem_bf, B_ * S_ * U_);

  // --- keys = memory @ W_k : [1920,512]x[512,512] ---
  gemm_tiled<<<dim3(2, (B_ * S_) / 64), 256, 64 * U_ * 2, stream>>>(
      mem_bf, Wk_t, keys, B_ * S_, U_, U_, U_, U_, nullptr, nullptr);

  // --- recurrent scan (16 WGs, 3 device syncs/step) ---
  (void)hipMemsetAsync(bar, 0, sizeof(unsigned), stream);
  scan_kernel<<<16, 1024, 0, stream>>>(xbf, Wz_t, Wa_t, bz, keys, mem, ih, ic,
                                       h_bf, ctx_bf, attn_bf, hs, cs, attns, attns_bf,
                                       aligns, bar);

  // --- pointer-generator head ---
  pgen_kernel<<<(B_ * T_ + 7) / 8, 256, 0, stream>>>(hs, cs, attns, dec, emb, Wg, bg, pgen);
  gemm_tiled<<<dim3((NFCP_ / 16 + 15) / 16, (B_ * T_) / 64), 256, 64 * U_ * 2, stream>>>(
      attns_bf, Wfc_t, out, B_ * T_, U_, V_, NFCP_, VEXT_, bfc, pgen);
  zero_oov<<<(B_ * T_ * OOV_ + 255) / 256, 256, 0, stream>>>(out);
  scatter_kernel<<<(B_ * T_ * S_ + 255) / 256, 256, 0, stream>>>(eids, aligns, pgen, out);
}
